// AttentionRNN_67190468378874
// MI455X (gfx1250) — matrix operand order repaired and checked by reading, not compiled, not run on hardware
//
#include <hip/hip_runtime.h>
#include <cstdint>
#include <cstddef>

// Problem sizes (match reference)
#define kB 128
#define kT 32
#define kD 1024
#define kH 1024
#define kV 2048
#define kA 512
#define kN 64

typedef __attribute__((ext_vector_type(16))) __bf16 bf16x16;
typedef __attribute__((ext_vector_type(8)))  float  f32x8;
typedef unsigned short u16;

// ---------------------------------------------------------------------------
// helpers
// ---------------------------------------------------------------------------
__device__ __forceinline__ u16 f2bf(float f) {
    unsigned int u = __float_as_uint(f);
    u += 0x7FFFu + ((u >> 16) & 1u);           // round-to-nearest-even
    return (u16)(u >> 16);
}
__device__ __forceinline__ float sigmoidf_(float x) {
    return 1.0f / (1.0f + expf(-x));
}

union ABFrag { bf16x16 v; uint4 q[2]; };

// A fragment (16x32 bf16, per CDNA5 ISA layout):
//   lanes 0-15 : row=lane,    K = k..k+7  (q0) and k+16..k+23 (q1)
//   lanes 16-31: row=lane-16, K = k+8..k+15 (q0) and k+24..k+31 (q1)
__device__ __forceinline__ bf16x16 load_a_frag(const u16* A, int rowbase, int ld,
                                               int k, int lane) {
    ABFrag u;
    int row  = rowbase + (lane & 15);
    int koff = (lane & 16) ? 8 : 0;
    const u16* p = A + (size_t)row * ld + k + koff;
    u.q[0] = *(const uint4*)(p);
    u.q[1] = *(const uint4*)(p + 16);
    return u.v;
}
// B fragment (32x16 bf16). W is [N,K] row-major so B^T(k,n)=W[n,k]:
//   lanes 0-15 : col=lane,    K = k..k+7  (q0) and k+16..k+23 (q1)
//   lanes 16-31: col=lane-16, K = k+8..k+15 (q0) and k+24..k+31 (q1)
__device__ __forceinline__ bf16x16 load_b_frag(const u16* W, int colbase, int ld,
                                               int k, int lane) {
    ABFrag u;
    int col = colbase + (lane & 15);
    int kb  = (lane & 16) ? 8 : 0;
    const u16* p = W + (size_t)col * ld + k + kb;
    u.q[0] = *(const uint4*)(p);
    u.q[1] = *(const uint4*)(p + 16);
    return u.v;
}

__device__ __forceinline__ f32x8 wmma_bf16(bf16x16 a, bf16x16 b, f32x8 c) {
    return __builtin_amdgcn_wmma_f32_16x16x32_bf16(
        /*neg_a=*/false, a, /*neg_b=*/false, b,
        /*c_mod=*/(short)0, c, /*reuse_a=*/false, /*reuse_b=*/false);
}

// load one pipeline stage: A-frag + 4 B-frags (issued as one clause)
__device__ __forceinline__ void load_stage(const u16* A, const u16* W, int K,
                                           int rowbase, int colbase, int k, int lane,
                                           bf16x16& a, bf16x16 b[4]) {
    a = load_a_frag(A, rowbase, K, k, lane);
#pragma unroll
    for (int j = 0; j < 4; ++j)
        b[j] = load_b_frag(W, colbase + 16 * j, K, k, lane);
}

// ---------------------------------------------------------------------------
// f32 -> bf16 conversion (grid-stride)
// ---------------------------------------------------------------------------
__global__ void k_f2bf(const float* __restrict__ in, u16* __restrict__ out, size_t n) {
    for (size_t i = (size_t)blockIdx.x * blockDim.x + threadIdx.x; i < n;
         i += (size_t)gridDim.x * blockDim.x)
        out[i] = f2bf(in[i]);
}

// ---------------------------------------------------------------------------
// masked mean over ROIs: mean_vis (bf16) [B,V]
// ---------------------------------------------------------------------------
__global__ __launch_bounds__(256) void k_meanvis(const float* __restrict__ visual,
                                                 const int* __restrict__ ls_rois,
                                                 u16* __restrict__ mvb) {
    int b = blockIdx.x;
    int nr = ls_rois[b];
    float inv = 1.0f / (float)nr;
    for (int v = threadIdx.x; v < kV; v += 256) {
        float s = 0.0f;
        for (int n = 0; n < nr; ++n)
            s += visual[((size_t)b * kN + n) * kV + v];
        mvb[(size_t)b * kV + v] = f2bf(s * inv);
    }
}

// ---------------------------------------------------------------------------
// WMMA bf16 GEMM:  C[M,Nc] = A1[M,K1] @ W1[Nc,K1]^T (+ A2[M,K2] @ W2[Nc,K2]^T)
//                           + bias1 (+ bias2)
// Wave tile 16x64 (A-frag reused 4x), block = 8 waves -> 32x256 tile.
// Software-pipelined (double-buffered fragments, two k-steps per loop body,
// copy-free ping-pong). Requires M%32==0, Nc%256==0, K%64==0 (true at every
// call site). Optional bf16 mirror of C for the next recurrent GEMM.
// ---------------------------------------------------------------------------
__global__ __launch_bounds__(256) void k_gemm_wmma(
    const u16* __restrict__ A1, const u16* __restrict__ W1, int K1,
    const u16* __restrict__ A2, const u16* __restrict__ W2, int K2,
    const float* __restrict__ bias1, const float* __restrict__ bias2,
    float* __restrict__ C, u16* __restrict__ Cbf, int M, int Nc) {

    const int tid   = threadIdx.x;
    const int lane  = tid & 31;
    const int wave  = tid >> 5;
    const int rowbase = blockIdx.y * 32  + (wave >> 2) * 16;
    const int colbase = blockIdx.x * 256 + (wave & 3) * 64;
    (void)M;

    f32x8 acc[4];
#pragma unroll
    for (int j = 0; j < 4; ++j)
#pragma unroll
        for (int r = 0; r < 8; ++r) acc[j][r] = 0.0f;

    const int wrow = colbase + (lane & 15);       // weight row this lane streams

    // ---- pass 1: A1 @ W1^T, double-buffered --------------------------------
    {
        bf16x16 a0, a1, b0[4], b1[4];
        load_stage(A1, W1, K1, rowbase, colbase, 0, lane, a0, b0);
        for (int k = 0; k < K1 - 64; k += 64) {
            __builtin_prefetch(W1 + (size_t)wrow * K1 + k + 128, 0, 3);
            load_stage(A1, W1, K1, rowbase, colbase, k + 32, lane, a1, b1);
#pragma unroll
            for (int j = 0; j < 4; ++j) acc[j] = wmma_bf16(a0, b0[j], acc[j]);
            load_stage(A1, W1, K1, rowbase, colbase, k + 64, lane, a0, b0);
#pragma unroll
            for (int j = 0; j < 4; ++j) acc[j] = wmma_bf16(a1, b1[j], acc[j]);
        }
        load_stage(A1, W1, K1, rowbase, colbase, K1 - 32, lane, a1, b1);
#pragma unroll
        for (int j = 0; j < 4; ++j) acc[j] = wmma_bf16(a0, b0[j], acc[j]);
#pragma unroll
        for (int j = 0; j < 4; ++j) acc[j] = wmma_bf16(a1, b1[j], acc[j]);
    }

    // ---- pass 2 (optional): A2 @ W2^T, accumulate into same tile -----------
    if (A2 != nullptr) {
        bf16x16 a0, a1, b0[4], b1[4];
        load_stage(A2, W2, K2, rowbase, colbase, 0, lane, a0, b0);
        for (int k = 0; k < K2 - 64; k += 64) {
            __builtin_prefetch(W2 + (size_t)wrow * K2 + k + 128, 0, 3);
            load_stage(A2, W2, K2, rowbase, colbase, k + 32, lane, a1, b1);
#pragma unroll
            for (int j = 0; j < 4; ++j) acc[j] = wmma_bf16(a0, b0[j], acc[j]);
            load_stage(A2, W2, K2, rowbase, colbase, k + 64, lane, a0, b0);
#pragma unroll
            for (int j = 0; j < 4; ++j) acc[j] = wmma_bf16(a1, b1[j], acc[j]);
        }
        load_stage(A2, W2, K2, rowbase, colbase, K2 - 32, lane, a1, b1);
#pragma unroll
        for (int j = 0; j < 4; ++j) acc[j] = wmma_bf16(a0, b0[j], acc[j]);
#pragma unroll
        for (int j = 0; j < 4; ++j) acc[j] = wmma_bf16(a1, b1[j], acc[j]);
    }

    // C/D layout: VGPR r holds M = r + (lane>=16 ? 8 : 0), N = lane&15
    const int mofs = (lane & 16) ? 8 : 0;
    const int ncol = lane & 15;
#pragma unroll
    for (int j = 0; j < 4; ++j) {
        int n = colbase + 16 * j + ncol;
        float bsum = bias1 ? bias1[n] : 0.0f;
        if (bias2) bsum += bias2[n];
#pragma unroll
        for (int r = 0; r < 8; ++r) {
            int m = rowbase + r + mofs;
            float val = acc[j][r] + bsum;
            C[(size_t)m * Nc + n] = val;
            if (Cbf) Cbf[(size_t)m * Nc + n] = f2bf(val);
        }
    }
}

// ---------------------------------------------------------------------------
// fused attention: logits = relu(va + he) @ Wp + bp, masked softmax over N,
// ctx = att @ visual. One block (8 waves) per batch row.
// ---------------------------------------------------------------------------
__global__ __launch_bounds__(256) void k_attention(
    const float* __restrict__ va, const float* __restrict__ he,
    const float* __restrict__ visual, const float* __restrict__ Wp,
    const float* __restrict__ bp, const int* __restrict__ ls_rois,
    float* __restrict__ ctx, u16* __restrict__ ctxb) {

    __shared__ float s_he[kA];
    __shared__ float s_logit[kN];
    __shared__ float s_att[kN];

    const int b = blockIdx.x;
    const int tid = threadIdx.x;
    const int lane = tid & 31;
    const int wave = tid >> 5;

    for (int i = tid; i < kA; i += 256) s_he[i] = he[(size_t)b * kA + i];
    __syncthreads();

    // logits: each wave handles 8 ROIs, lanes parallel over A=512
    for (int n = wave; n < kN; n += 8) {
        const float* var = va + ((size_t)b * kN + n) * kA;
        float s = 0.0f;
        for (int a = lane; a < kA; a += 32) {
            float v = var[a] + s_he[a];
            v = v > 0.0f ? v : 0.0f;
            s += v * Wp[a];
        }
#pragma unroll
        for (int off = 16; off; off >>= 1) s += __shfl_xor(s, off, 32);
        if (lane == 0) s_logit[n] = s + bp[0];
    }
    __syncthreads();

    // masked softmax over 64 entries (tiny; single thread)
    if (tid == 0) {
        int nr = ls_rois[b];
        float mx = -1e30f;
        for (int n = 0; n < nr; ++n) mx = fmaxf(mx, s_logit[n]);
        float sum = 0.0f;
        for (int n = 0; n < kN; ++n) {
            float e = (n < nr) ? expf(s_logit[n] - mx) : 0.0f;
            s_att[n] = e;
            sum += e;
        }
        float inv = 1.0f / sum;
        for (int n = 0; n < kN; ++n) s_att[n] *= inv;
    }
    __syncthreads();

    // ctx = sum_n att[n] * visual[b,n,:]
    for (int v = tid; v < kV; v += 256) {
        float s = 0.0f;
        const float* vis = visual + (size_t)b * kN * kV + v;
#pragma unroll 4
        for (int n = 0; n < kN; ++n) s += s_att[n] * vis[(size_t)n * kV];
        ctx[(size_t)b * kV + v]  = s;
        ctxb[(size_t)b * kV + v] = f2bf(s);
    }
}

// ---------------------------------------------------------------------------
// fused LSTM pointwise: gates + highway + seq-mask + y = out@Wo + bo,
// writes h (f32 + bf16 mirror) and c. One block per batch row.
// ---------------------------------------------------------------------------
__global__ __launch_bounds__(256) void k_lstm_pointwise(
    const float* __restrict__ pi, const float* __restrict__ pre,
    float* __restrict__ h, float* __restrict__ c, u16* __restrict__ hb,
    const float* __restrict__ Wo, const float* __restrict__ bo,
    const int* __restrict__ seq_lens, float* __restrict__ y, int t) {

    const int b = blockIdx.x;
    const int tid = threadIdx.x;
    const int lane = tid & 31;
    const int wave = tid >> 5;
    const bool valid = t < seq_lens[b];

    const float* pir = pi  + ((size_t)b * kT + t) * (6 * kH);
    const float* prr = pre + (size_t)b * (5 * kH);
    float part = 0.0f;

    for (int j = tid; j < kH; j += 256) {
        size_t hidx = (size_t)b * kH + j;
        float i_g = sigmoidf_(pir[j]          + prr[j]);
        float f_g = sigmoidf_(pir[kH + j]     + prr[kH + j]);
        float m_i = tanhf    (pir[2 * kH + j] + prr[2 * kH + j]);
        float o_g = sigmoidf_(pir[3 * kH + j] + prr[3 * kH + j]);
        float mem = i_g * m_i + f_g * c[hidx];
        float out = o_g * tanhf(mem);
        float hw  = sigmoidf_(pir[4 * kH + j] + prr[4 * kH + j]);
        out = hw * out + (1.0f - hw) * pir[5 * kH + j];
        float hn = valid ? out : h[hidx];
        float cn = valid ? mem : c[hidx];
        h[hidx] = hn;
        c[hidx] = cn;
        hb[hidx] = f2bf(hn);
        part += out * Wo[j];
    }
#pragma unroll
    for (int off = 16; off; off >>= 1) part += __shfl_xor(part, off, 32);
    __shared__ float s_red[8];
    if (lane == 0) s_red[wave] = part;
    __syncthreads();
    if (tid == 0) {
        float tot = 0.0f;
        for (int i = 0; i < 8; ++i) tot += s_red[i];
        y[(size_t)b * kT + t] = valid ? (tot + bo[0]) : 0.0f;
    }
}

// ---------------------------------------------------------------------------
// host
// ---------------------------------------------------------------------------
extern "C" void kernel_launch(void* const* d_in, const int* in_sizes, int n_in,
                              void* d_out, int out_size, void* d_ws, size_t ws_size,
                              hipStream_t stream) {
    (void)in_sizes; (void)n_in; (void)out_size; (void)ws_size;

    const float* x      = (const float*)d_in[0];
    const float* visual = (const float*)d_in[1];
    const float* Wi     = (const float*)d_in[2];
    const float* bi     = (const float*)d_in[3];
    const float* Ws     = (const float*)d_in[4];
    const float* bs     = (const float*)d_in[5];
    const float* Wa     = (const float*)d_in[6];
    const float* ba     = (const float*)d_in[7];
    const float* Wv     = (const float*)d_in[8];
    const float* bv     = (const float*)d_in[9];
    const float* Wh     = (const float*)d_in[10];
    const float* bh     = (const float*)d_in[11];
    const float* Wp     = (const float*)d_in[12];
    const float* bp     = (const float*)d_in[13];
    const float* W0h    = (const float*)d_in[14];
    const float* b0h    = (const float*)d_in[15];
    const float* W0c    = (const float*)d_in[16];
    const float* b0c    = (const float*)d_in[17];
    const float* Wo     = (const float*)d_in[18];
    const float* bo     = (const float*)d_in[19];
    const int* ls_rois  = (const int*)d_in[20];
    const int* seq_lens = (const int*)d_in[21];
    float* y = (float*)d_out;

    // workspace carve-out
    char* wp = (char*)d_ws;
    auto alloc = [&](size_t bytes) -> void* {
        void* p = (void*)wp;
        wp += (bytes + 255) & ~(size_t)255;
        return p;
    };
    u16* xb   = (u16*)alloc((size_t)kB * kT * kD * 2);
    u16* visb = (u16*)alloc((size_t)kB * kN * kV * 2);
    u16* Wib  = (u16*)alloc((size_t)6 * kH * kD * 2);
    u16* Wsb  = (u16*)alloc((size_t)5 * kH * kH * 2);
    u16* Wab  = (u16*)alloc((size_t)5 * kH * kV * 2);
    u16* Wvb  = (u16*)alloc((size_t)kA * kV * 2);
    u16* Whb  = (u16*)alloc((size_t)kA * kH * 2);
    u16* W0hb = (u16*)alloc((size_t)kH * kV * 2);
    u16* W0cb = (u16*)alloc((size_t)kH * kV * 2);
    u16* mvb  = (u16*)alloc((size_t)kB * kV * 2);
    u16* hb   = (u16*)alloc((size_t)kB * kH * 2);
    u16* ctxb = (u16*)alloc((size_t)kB * kV * 2);
    float* h    = (float*)alloc((size_t)kB * kH * 4);
    float* c    = (float*)alloc((size_t)kB * kH * 4);
    float* va   = (float*)alloc((size_t)kB * kN * kA * 4);
    float* pi   = (float*)alloc((size_t)kB * kT * 6 * kH * 4);
    float* he   = (float*)alloc((size_t)kB * kA * 4);
    float* pre  = (float*)alloc((size_t)kB * 5 * kH * 4);
    float* ctx  = (float*)alloc((size_t)kB * kV * 4);

    // 1) one-time bf16 conversions
    auto conv = [&](const float* src, u16* dst, size_t n) {
        k_f2bf<<<4096, 256, 0, stream>>>(src, dst, n);
    };
    conv(x,      xb,   (size_t)kB * kT * kD);
    conv(visual, visb, (size_t)kB * kN * kV);
    conv(Wi,  Wib,  (size_t)6 * kH * kD);
    conv(Ws,  Wsb,  (size_t)5 * kH * kH);
    conv(Wa,  Wab,  (size_t)5 * kH * kV);
    conv(Wv,  Wvb,  (size_t)kA * kV);
    conv(Wh,  Whb,  (size_t)kA * kH);
    conv(W0h, W0hb, (size_t)kH * kV);
    conv(W0c, W0cb, (size_t)kH * kV);

    // 2) mean_vis (bf16)
    k_meanvis<<<kB, 256, 0, stream>>>(visual, ls_rois, mvb);

    // 3) loop-invariant GEMMs
    // h0 = mean_vis @ W0h^T + b0h   (also bf16 mirror for recurrence)
    k_gemm_wmma<<<dim3(kH / 256, kB / 32), 256, 0, stream>>>(
        mvb, W0hb, kV, nullptr, nullptr, 0, b0h, nullptr, h, hb, kB, kH);
    // c0 = mean_vis @ W0c^T + b0c
    k_gemm_wmma<<<dim3(kH / 256, kB / 32), 256, 0, stream>>>(
        mvb, W0cb, kV, nullptr, nullptr, 0, b0c, nullptr, c, nullptr, kB, kH);
    // va = visual @ Wv^T + bv      [B*N, A]
    k_gemm_wmma<<<dim3(kA / 256, (kB * kN) / 32), 256, 0, stream>>>(
        visb, Wvb, kV, nullptr, nullptr, 0, bv, nullptr, va, nullptr, kB * kN, kA);
    // pi = x @ Wi^T + bi           [B*T, 6H]
    k_gemm_wmma<<<dim3((6 * kH) / 256, (kB * kT) / 32), 256, 0, stream>>>(
        xb, Wib, kD, nullptr, nullptr, 0, bi, nullptr, pi, nullptr, kB * kT, 6 * kH);

    // 4) recurrence
    for (int t = 0; t < kT; ++t) {
        // he = h @ Wh^T + bh       [B, A]
        k_gemm_wmma<<<dim3(kA / 256, kB / 32), 256, 0, stream>>>(
            hb, Whb, kH, nullptr, nullptr, 0, bh, nullptr, he, nullptr, kB, kA);
        // attention + context
        k_attention<<<kB, 256, 0, stream>>>(va, he, visual, Wp, bp, ls_rois, ctx, ctxb);
        // pre = h @ Ws^T + ctx @ Wa^T + bs + ba   [B, 5H]  (dual-K fused)
        k_gemm_wmma<<<dim3((5 * kH) / 256, kB / 32), 256, 0, stream>>>(
            hb, Wsb, kH, ctxb, Wab, kV, bs, ba, pre, nullptr, kB, 5 * kH);
        // gates, state update, y
        k_lstm_pointwise<<<kB, 256, 0, stream>>>(pi, pre, h, c, hb, Wo, bo,
                                                 seq_lens, y, t);
    }
}
